// MultiheadAttention_55954833932873
// MI455X (gfx1250) — matrix operand order repaired and checked by reading, not compiled, not run on hardware
//
#include <hip/hip_runtime.h>

typedef __attribute__((ext_vector_type(16))) _Float16 v16h;
typedef __attribute__((ext_vector_type(8)))  _Float16 v8h;
typedef __attribute__((ext_vector_type(4)))  _Float16 v4h;
typedef __attribute__((ext_vector_type(8)))  float    v8f;
typedef __attribute__((ext_vector_type(4)))  float    v4f;
typedef __attribute__((ext_vector_type(4)))  unsigned int u32x4;
typedef __attribute__((ext_vector_type(8)))  int      i32x8;
typedef __attribute__((ext_vector_type(4)))  int      i32x4;

#define D_MODEL 1024
#define N_HEAD  16
#define D_HEAD  64
#define BATCH   2
#define SEQ     2048
#define MROWS   (BATCH * SEQ)   // 4096 total rows

// ---- CDNA5 feature detection (compile-time, per-pass) ----------------------
#if defined(__has_builtin)
#if __has_builtin(__builtin_amdgcn_tensor_load_to_lds) && \
    __has_builtin(__builtin_amdgcn_s_wait_tensorcnt)
#define HAVE_TDM 1
#endif
#if __has_builtin(__builtin_amdgcn_global_load_async_to_lds_b128) && \
    __has_builtin(__builtin_amdgcn_s_wait_asynccnt)
#define HAVE_ASYNC 1
#endif
#endif
#if __has_include(<hip/amd_detail/amd_gfx1250_TDM.h>)
#define TDM_ARITY6 1   // therock-10.0 headers => 6-arg builtin
#endif

// LDS byte offset of a generic pointer into a __shared__ array.
__device__ __forceinline__ unsigned lds_off(const void* p) {
    typedef __attribute__((address_space(3))) const void lds_cv;
    return (unsigned)(unsigned long long)(lds_cv*)p;
}

#if defined(HAVE_TDM)
// TDM 2-D f16 tile load: copy tile_h rows x tile_w elems (row stride
// row_stride elems) from gsrc into contiguous LDS at ldsOff.
// Descriptor per CDNA5 ISA ch.8 (D# groups 0/1; groups 2/3 zero => 2-D).
__device__ __forceinline__ void tdm_load_tile_f16(const _Float16* gsrc,
                                                  unsigned ldsOff,
                                                  unsigned tensor_w,
                                                  unsigned tensor_h,
                                                  unsigned tile_w,
                                                  unsigned tile_h,
                                                  unsigned row_stride) {
    unsigned long long ga = (unsigned long long)gsrc;
    u32x4 g0;
    g0[0] = 1u;                                        // count=1, user desc
    g0[1] = ldsOff;                                    // lds_addr (bytes)
    g0[2] = (unsigned)(ga & 0xFFFFFFFFull);            // global_addr[31:0]
    g0[3] = (unsigned)((ga >> 32) & 0x01FFFFFFull)     // global_addr[56:32]
            | 0x80000000u;                             // type=2 (bits 127:126)
    i32x8 g1;
    g1[0] = (int)(1u << 16);                           // data_size=1 (2 bytes)
    g1[1] = (int)((tensor_w & 0xFFFFu) << 16);         // tensor_dim0 lo16
    g1[2] = (int)((tensor_w >> 16) |
                  ((tensor_h & 0xFFFFu) << 16));       // dim0 hi16 | dim1 lo16
    g1[3] = (int)((tensor_h >> 16) |
                  ((tile_w & 0xFFFFu) << 16));         // dim1 hi16 | tile_dim0
    g1[4] = (int)(tile_h & 0xFFFFu);                   // tile_dim1 (tile_dim2=0)
    g1[5] = (int)row_stride;                           // tensor_dim0_stride lo32
    g1[6] = 0;                                         // stride hi16 | dim1_stride
    g1[7] = 0;
    i32x4 z4; z4[0] = 0; z4[1] = 0; z4[2] = 0; z4[3] = 0;
#if defined(TDM_ARITY6)
    i32x8 z8; 
#pragma unroll
    for (int i = 0; i < 8; ++i) z8[i] = 0;
    __builtin_amdgcn_tensor_load_to_lds(g0, g1, z4, z4, z8, 0);
#else
    __builtin_amdgcn_tensor_load_to_lds(g0, g1, z4, z4, 0);
#endif
}
#endif

// ---------------------------------------------------------------------------
// WMMA helper: D = A(16x32 f16) * B(32x16 f16) + C(f32)
// ---------------------------------------------------------------------------
__device__ __forceinline__ v8f wmma_f16(v16h a, v16h b, v8f c) {
    return __builtin_amdgcn_wmma_f32_16x16x32_f16(
        false, a, false, b, (short)0, c, false, false);
}

// A-operand fragment (16x32, MxK) from row-major [row][k] tile.
__device__ __forceinline__ v16h ld_frag_a(const _Float16* base, int row16,
                                          int stride, int lane) {
    int r    = row16 + (lane & 15);
    int koff = (lane >> 4) << 3;
    const _Float16* p = base + r * stride + koff;
    v8h lo = *(const v8h*)(p);
    v8h hh = *(const v8h*)(p + 16);
    v16h out;
#pragma unroll
    for (int i = 0; i < 8; ++i) { out[i] = lo[i]; out[i + 8] = hh[i]; }
    return out;
}

// B-operand fragment (32x16, KxN) from row-major [n][k] tile.
__device__ __forceinline__ v16h ld_frag_b(const _Float16* base, int row16,
                                          int stride, int lane) {
    int r    = row16 + (lane & 15);
    int koff = (lane >> 4) << 3;
    const _Float16* p = base + r * stride + koff;
    v8h lo = *(const v8h*)(p);
    v8h hh = *(const v8h*)(p + 16);
    v16h out;
#pragma unroll
    for (int i = 0; i < 8; ++i) { out[i] = lo[i]; out[i + 8] = hh[i]; }
    return out;
}

__device__ __forceinline__ float half_max16(float v) {
#pragma unroll
    for (int off = 1; off < 16; off <<= 1) v = fmaxf(v, __shfl_xor(v, off, 32));
    return v;
}
__device__ __forceinline__ float half_sum16(float v) {
#pragma unroll
    for (int off = 1; off < 16; off <<= 1) v += __shfl_xor(v, off, 32);
    return v;
}

// ---------------------------------------------------------------------------
// fp32 -> f16 conversion (weights)
// ---------------------------------------------------------------------------
__global__ void cvt_f32_f16_kernel(const float* __restrict__ in,
                                   _Float16* __restrict__ out, int n) {
    int idx = (blockIdx.x * blockDim.x + threadIdx.x) * 4;
    if (idx < n) {
        v4f f = *(const v4f*)(in + idx);
        v4h h;
#pragma unroll
        for (int i = 0; i < 4; ++i) h[i] = (_Float16)f[i];
        *(v4h*)(out + idx) = h;
    }
}

// ---------------------------------------------------------------------------
// GEMM: C[M,N] = A[M,K](fp32) * W[N,K](f16)^T + bias[N]
// 128x128x32 tiles, 8 waves/block. Weight tile staged by TDM when available.
// ---------------------------------------------------------------------------
__global__ __launch_bounds__(256)
void gemm_xWt_kernel(const float* __restrict__ A, const _Float16* __restrict__ W,
                     const float* __restrict__ bias,
                     float* __restrict__ Cf, _Float16* __restrict__ Ch,
                     int M, int N, int K) {
    __shared__ _Float16 sA[128 * 32];   // 8 KB
    __shared__ _Float16 sB[128 * 32];   // 8 KB

    const int tid  = threadIdx.x;
    const int lane = tid & 31;
    const int wave = tid >> 5;          // 0..7
    const int wm   = wave >> 1;         // 0..3 : 32-row group
    const int wn   = wave & 1;          // 0..1 : 64-col group
    const int bM   = blockIdx.y * 128;
    const int bN   = blockIdx.x * 128;

    v8f acc[2][4];
#pragma unroll
    for (int i = 0; i < 2; ++i)
#pragma unroll
        for (int j = 0; j < 4; ++j)
#pragma unroll
            for (int e = 0; e < 8; ++e) acc[i][j][e] = 0.0f;

    const int srow = tid >> 1;          // 0..127
    const int skp  = (tid & 1) * 16;    // 0 or 16

    for (int k0 = 0; k0 < K; k0 += 32) {
        __syncthreads();
#if defined(HAVE_TDM)
        // Weight tile 128 rows x 32 f16, strided by K, via Tensor Data Mover.
        if (wave == 0) {
            tdm_load_tile_f16(W + (size_t)bN * K + k0, lds_off(sB),
                              (unsigned)K, (unsigned)N, 32u, 128u, (unsigned)K);
        }
#endif
        // Stage A tile (fp32 -> f16), all threads
        {
            const float* ga = A + (size_t)(bM + srow) * K + k0 + skp;
            v4f f0 = *(const v4f*)(ga);
            v4f f1 = *(const v4f*)(ga + 4);
            v4f f2 = *(const v4f*)(ga + 8);
            v4f f3 = *(const v4f*)(ga + 12);
            v8h h0, h1;
#pragma unroll
            for (int i = 0; i < 4; ++i) {
                h0[i] = (_Float16)f0[i]; h0[i + 4] = (_Float16)f1[i];
                h1[i] = (_Float16)f2[i]; h1[i + 4] = (_Float16)f3[i];
            }
            *(v8h*)&sA[srow * 32 + skp]     = h0;
            *(v8h*)&sA[srow * 32 + skp + 8] = h1;
        }
#if defined(HAVE_TDM)
        if (wave == 0) __builtin_amdgcn_s_wait_tensorcnt(0);
#elif defined(HAVE_ASYNC)
        {
            typedef __attribute__((address_space(1))) const void gv;
            typedef __attribute__((address_space(3))) void lv;
            const _Float16* gw = W + (size_t)(bN + srow) * K + k0 + skp;
            __builtin_amdgcn_global_load_async_to_lds_b128(
                (gv*)gw, (lv*)&sB[srow * 32 + skp], 0, 0);
            __builtin_amdgcn_global_load_async_to_lds_b128(
                (gv*)(gw + 8), (lv*)&sB[srow * 32 + skp + 8], 0, 0);
        }
        __builtin_amdgcn_s_wait_asynccnt(0);
#else
        {
            const _Float16* gw = W + (size_t)(bN + srow) * K + k0 + skp;
            *(v8h*)&sB[srow * 32 + skp]     = *(const v8h*)(gw);
            *(v8h*)&sB[srow * 32 + skp + 8] = *(const v8h*)(gw + 8);
        }
#endif
        __syncthreads();

        v16h af0 = ld_frag_a(sA, wm * 32,      32, lane);
        v16h af1 = ld_frag_a(sA, wm * 32 + 16, 32, lane);
#pragma unroll
        for (int j = 0; j < 4; ++j) {
            v16h bf = ld_frag_b(sB, wn * 64 + j * 16, 32, lane);
            acc[0][j] = wmma_f16(af0, bf, acc[0][j]);
            acc[1][j] = wmma_f16(af1, bf, acc[1][j]);
        }
    }

    const int hi = (lane >> 4) << 3;
#pragma unroll
    for (int j = 0; j < 4; ++j) {
        int n = bN + wn * 64 + j * 16 + (lane & 15);
        float bn = bias[n];
#pragma unroll
        for (int i = 0; i < 2; ++i) {
#pragma unroll
            for (int r = 0; r < 8; ++r) {
                int m = bM + wm * 32 + i * 16 + r + hi;
                float val = acc[i][j][r] + bn;
                if (Cf) Cf[(size_t)m * N + n] = val;
                else    Ch[(size_t)m * N + n] = (_Float16)val;
            }
        }
    }
}

// ---------------------------------------------------------------------------
// Flash attention: per (b, h, 64 q rows). 4 waves x 16 q rows each.
// K tile staged by TDM when available; V transposed manually.
// ---------------------------------------------------------------------------
__global__ __launch_bounds__(128)
void flash_attn_kernel(const _Float16* __restrict__ Q,
                       const _Float16* __restrict__ Km,
                       const _Float16* __restrict__ V,
                       float* __restrict__ AO) {
    const int qb   = blockIdx.x;
    const int bh   = blockIdx.y;
    const int b    = bh >> 4;
    const int h    = bh & 15;
    const int tid  = threadIdx.x;
    const int lane = tid & 31;
    const int wave = tid >> 5;              // 0..3
    const int qRow = qb * 64 + wave * 16;
    const int hi   = (lane >> 4) << 3;

    __shared__ _Float16 sK[32 * 64];        // [kv][d]   4 KB
    __shared__ _Float16 sVt[64 * 32];       // [d][kv]   4 KB
    __shared__ _Float16 sP[4 * 16 * 32];    // per-wave P patch, 4 KB

    const _Float16* qbase = Q + ((size_t)b * SEQ) * D_MODEL + h * D_HEAD;

    v16h qf0, qf1;
    {
        int r    = qRow + (lane & 15);
        int koff = (lane >> 4) << 3;
        const _Float16* p = qbase + (size_t)r * D_MODEL + koff;
        v8h a0 = *(const v8h*)(p);      v8h a1 = *(const v8h*)(p + 16);
        v8h a2 = *(const v8h*)(p + 32); v8h a3 = *(const v8h*)(p + 48);
#pragma unroll
        for (int i = 0; i < 8; ++i) {
            qf0[i] = a0[i]; qf0[i + 8] = a1[i];
            qf1[i] = a2[i]; qf1[i + 8] = a3[i];
        }
    }

    v8f o[4];
#pragma unroll
    for (int j = 0; j < 4; ++j)
#pragma unroll
        for (int e = 0; e < 8; ++e) o[j][e] = 0.0f;

    float mrow[8], lrow[8];
#pragma unroll
    for (int r = 0; r < 8; ++r) { mrow[r] = -3.0e38f; lrow[r] = 0.0f; }

    const float SCL   = 0.125f;             // 1/sqrt(64)
    const float LOG2E = 1.4426950408889634f;
    _Float16* pw = sP + wave * (16 * 32);

    const int srow = tid >> 2;              // 0..31 kv row
    const int sdp  = (tid & 3) * 16;        // d sub-chunk

    for (int kv0 = 0; kv0 < SEQ; kv0 += 32) {
        __syncthreads();
#if defined(HAVE_TDM)
        if (wave == 0) {
            tdm_load_tile_f16(
                Km + ((size_t)b * SEQ + kv0) * D_MODEL + h * D_HEAD,
                lds_off(sK), (unsigned)D_MODEL, (unsigned)MROWS,
                64u, 32u, (unsigned)D_MODEL);
        }
#else
        {
            const _Float16* gk =
                Km + ((size_t)b * SEQ + kv0 + srow) * D_MODEL + h * D_HEAD + sdp;
            *(v8h*)&sK[srow * 64 + sdp]     = *(const v8h*)(gk);
            *(v8h*)&sK[srow * 64 + sdp + 8] = *(const v8h*)(gk + 8);
        }
#endif
        // Stage V transposed [d][kv] (all threads)
        {
            const _Float16* gv =
                V + ((size_t)b * SEQ + kv0 + srow) * D_MODEL + h * D_HEAD + sdp;
            v8h v0 = *(const v8h*)(gv);
            v8h v1 = *(const v8h*)(gv + 8);
#pragma unroll
            for (int i = 0; i < 8; ++i) {
                sVt[(sdp + i) * 32 + srow]     = v0[i];
                sVt[(sdp + 8 + i) * 32 + srow] = v1[i];
            }
        }
#if defined(HAVE_TDM)
        if (wave == 0) __builtin_amdgcn_s_wait_tensorcnt(0);
#endif
        __syncthreads();

        v8f sc0, sc1;
#pragma unroll
        for (int e = 0; e < 8; ++e) { sc0[e] = 0.0f; sc1[e] = 0.0f; }
        {
            v16h b00 = ld_frag_b(sK,      0,  64, lane);
            v16h b01 = ld_frag_b(sK + 32, 0,  64, lane);
            sc0 = wmma_f16(qf0, b00, sc0);
            sc0 = wmma_f16(qf1, b01, sc0);
            v16h b10 = ld_frag_b(sK,      16, 64, lane);
            v16h b11 = ld_frag_b(sK + 32, 16, 64, lane);
            sc1 = wmma_f16(qf0, b10, sc1);
            sc1 = wmma_f16(qf1, b11, sc1);
        }

#pragma unroll
        for (int r = 0; r < 8; ++r) {
            float a = sc0[r] * SCL;
            float c = sc1[r] * SCL;
            float mt = half_max16(fmaxf(a, c));
            float mn = fmaxf(mrow[r], mt);
            float p0 = exp2f((a - mn) * LOG2E);
            float p1 = exp2f((c - mn) * LOG2E);
            float cf = exp2f((mrow[r] - mn) * LOG2E);
            mrow[r] = mn;
            lrow[r] = lrow[r] * cf + half_sum16(p0 + p1);
#pragma unroll
            for (int j = 0; j < 4; ++j) o[j][r] *= cf;
            pw[(r + hi) * 32 + (lane & 15)]      = (_Float16)p0;
            pw[(r + hi) * 32 + 16 + (lane & 15)] = (_Float16)p1;
        }

        v16h pf = ld_frag_a(pw, 0, 32, lane);
#pragma unroll
        for (int j = 0; j < 4; ++j) {
            v16h vb = ld_frag_b(sVt, j * 16, 32, lane);
            o[j] = wmma_f16(pf, vb, o[j]);
        }
    }

#pragma unroll
    for (int r = 0; r < 8; ++r) {
        int m = qRow + r + hi;
        float inv = 1.0f / lrow[r];
        float* dst = AO + ((size_t)b * SEQ + m) * D_MODEL + h * D_HEAD + (lane & 15);
#pragma unroll
        for (int j = 0; j < 4; ++j) dst[j * 16] = o[j][r] * inv;
    }
}

// ---------------------------------------------------------------------------
// Host-side launcher
// ---------------------------------------------------------------------------
extern "C" void kernel_launch(void* const* d_in, const int* in_sizes, int n_in,
                              void* d_out, int out_size, void* d_ws, size_t ws_size,
                              hipStream_t stream) {
    (void)in_sizes; (void)n_in; (void)out_size; (void)ws_size;
    const float* q  = (const float*)d_in[0];
    const float* k  = (const float*)d_in[1];
    const float* v  = (const float*)d_in[2];
    const float* Wq = (const float*)d_in[3];
    const float* bq = (const float*)d_in[4];
    const float* Wk = (const float*)d_in[5];
    const float* bk = (const float*)d_in[6];
    const float* Wv = (const float*)d_in[7];
    const float* bv = (const float*)d_in[8];
    const float* Wo = (const float*)d_in[9];
    const float* bo = (const float*)d_in[10];
    float* out = (float*)d_out;

    const size_t WE = (size_t)D_MODEL * D_MODEL;  // 1 Mi elems per weight
    const size_t AE = (size_t)MROWS * D_MODEL;    // 4 Mi elems per activation

    _Float16* hws  = (_Float16*)d_ws;
    _Float16* wq16 = hws;
    _Float16* wk16 = hws + WE;
    _Float16* wv16 = hws + 2 * WE;
    _Float16* wo16 = hws + 3 * WE;
    _Float16* q16  = hws + 4 * WE;
    _Float16* k16  = q16 + AE;
    _Float16* v16  = k16 + AE;
    float*    ao   = (float*)(v16 + AE);          // fp32 attention output

    {
        int n = (int)WE;
        int blocks = n / (256 * 4);
        cvt_f32_f16_kernel<<<blocks, 256, 0, stream>>>(Wq, wq16, n);
        cvt_f32_f16_kernel<<<blocks, 256, 0, stream>>>(Wk, wk16, n);
        cvt_f32_f16_kernel<<<blocks, 256, 0, stream>>>(Wv, wv16, n);
        cvt_f32_f16_kernel<<<blocks, 256, 0, stream>>>(Wo, wo16, n);
    }

    dim3 gg(D_MODEL / 128, MROWS / 128);          // (8, 32)
    gemm_xWt_kernel<<<gg, 256, 0, stream>>>(q, wq16, bq, nullptr, q16,
                                            MROWS, D_MODEL, D_MODEL);
    gemm_xWt_kernel<<<gg, 256, 0, stream>>>(k, wk16, bk, nullptr, k16,
                                            MROWS, D_MODEL, D_MODEL);
    gemm_xWt_kernel<<<gg, 256, 0, stream>>>(v, wv16, bv, nullptr, v16,
                                            MROWS, D_MODEL, D_MODEL);

    dim3 fg(SEQ / 64, BATCH * N_HEAD);            // (32, 32)
    flash_attn_kernel<<<fg, 128, 0, stream>>>(q16, k16, v16, ao);

    gemm_xWt_kernel<<<gg, 256, 0, stream>>>(ao, wo16, bo, out, nullptr,
                                            MROWS, D_MODEL, D_MODEL);
}
